// MultiHeadAttention_74010876444993
// MI455X (gfx1250) — matrix operand order repaired and checked by reading, not compiled, not run on hardware
//
#include <hip/hip_runtime.h>
#include <cstdint>
#include <cstddef>

// ---------------------------------------------------------------------------
// MHA for MI455X (gfx1250): bf16 WMMA (16x16x32, f32 accum) everywhere.
//   K1: fused per-head QKV projection GEMMs (pipelined, fp32->bf16 native cvt)
//   K2: flash attention (QK^T, online softmax, PV); async K/Q tile loads
//   K3: output projection ctx @ Wo^T + bo (pipelined, async A tile)
// ---------------------------------------------------------------------------

typedef __attribute__((ext_vector_type(16))) __bf16        v16bf;
typedef __attribute__((ext_vector_type(8)))  float         v8f;
typedef __attribute__((ext_vector_type(8)))  unsigned int  v8u;
typedef __attribute__((ext_vector_type(4)))  int           v4i;

constexpr int B_  = 4, S_ = 2048, E_ = 1024, H_ = 16, Dh_ = 64;
constexpr int BS_ = B_ * S_;  // 8192

#if __has_builtin(__builtin_amdgcn_global_load_async_to_lds_b128) && \
    __has_builtin(__builtin_amdgcn_s_wait_asynccnt)
#define HAS_ASYNC 1
#else
#define HAS_ASYNC 0
#endif

// async 16B copy global -> LDS (CDNA5 GLOBAL_LOAD_ASYNC_TO_LDS_B128);
// fallback: synchronous copy through VGPRs.
__device__ __forceinline__ void async_b128(void* lds, const void* g) {
#if HAS_ASYNC
  __builtin_amdgcn_global_load_async_to_lds_b128((v4i*)g, (v4i*)lds, 0, 0);
#else
  *reinterpret_cast<uint4*>(lds) = *reinterpret_cast<const uint4*>(g);
#endif
}
__device__ __forceinline__ void wait_async() {
#if HAS_ASYNC
  __builtin_amdgcn_s_wait_asynccnt(0);
#endif
}

// float -> bf16 via native converter (v_cvt_*bf16*), RNE
__device__ __forceinline__ unsigned short f2bf(float f) {
  return __builtin_bit_cast(unsigned short, (__bf16)f);
}

__device__ __forceinline__ v16bf make_frag(uint4 lo, uint4 hi) {
  v8u t;
  t[0] = lo.x; t[1] = lo.y; t[2] = lo.z; t[3] = lo.w;
  t[4] = hi.x; t[5] = hi.y; t[6] = hi.z; t[7] = hi.w;
  return __builtin_bit_cast(v16bf, t);
}

// A operand (16x32 bf16) from LDS tile stored row-major, row stride ldk elems.
__device__ __forceinline__ v16bf load_A(const unsigned short* lds, int lane,
                                        int ldk, int k0) {
  int row = lane & 15, hl = lane >> 4;
  const unsigned short* p = lds + row * ldk + k0 + hl * 8;
  uint4 lo = *reinterpret_cast<const uint4*>(p);
  uint4 hi = *reinterpret_cast<const uint4*>(p + 16);
  return make_frag(lo, hi);
}

// B operand (32K x 16N bf16) from LDS tile stored N-major [n][k], stride ldk.
__device__ __forceinline__ v16bf load_B(const unsigned short* lds, int lane,
                                        int ldk, int n0, int k0) {
  int n = n0 + (lane & 15), hl = lane >> 4;
  const unsigned short* p = lds + n * ldk + k0 + hl * 8;
  uint4 lo = *reinterpret_cast<const uint4*>(p);
  uint4 hi = *reinterpret_cast<const uint4*>(p + 16);
  return make_frag(lo, hi);
}

#define WMMA_BF16(a, b, c) \
  __builtin_amdgcn_wmma_f32_16x16x32_bf16(false, (a), false, (b), (short)0, (c), false, false)

// ---------------------------------------------------------------------------
// Kernel 1: per-head QKV projection. grid = (BS/128, 3*H), block = 256.
// 128x64 output tile; K-loop over E in 64-wide chunks, double-buffered LDS,
// next chunk's global loads issued before current chunk's WMMAs.
// ---------------------------------------------------------------------------
__global__ __launch_bounds__(256) void qkv_proj_kernel(
    const float* __restrict__ qx, const float* __restrict__ kx,
    const float* __restrict__ vx, const float* __restrict__ Wq,
    const float* __restrict__ Wk, const float* __restrict__ Wv,
    const float* __restrict__ bq, const float* __restrict__ bk,
    const float* __restrict__ bv, unsigned short* __restrict__ Qp,
    unsigned short* __restrict__ Kp, unsigned short* __restrict__ Vp) {
  __shared__ unsigned short aT[2][128 * 64];  // activations [m][k]
  __shared__ unsigned short bT[2][64 * 64];   // weights, N-major [d][k]

  const int tid = threadIdx.x;
  const int lane = tid & 31, wave = tid >> 5;
  const int m0 = blockIdx.x * 128;
  const int mat = blockIdx.y / H_, h = blockIdx.y % H_;

  const float* X = (mat == 0) ? qx : (mat == 1) ? kx : vx;
  const float* W = ((mat == 0) ? Wq : (mat == 1) ? Wk : Wv) + (size_t)h * E_ * Dh_;
  const float* bias = ((mat == 0) ? bq : (mat == 1) ? bk : bv) + h * Dh_;
  unsigned short* Out = (mat == 0) ? Qp : (mat == 1) ? Kp : Vp;

  const int kq = (tid & 15) * 4;   // A-tile k group
  const int ar0 = tid >> 4;        // A-tile row base (+ i*16)
  const int d0 = (tid & 15) * 4;   // B-tile d group
  const int br0 = tid >> 4;        // B-tile k base (+ i*16)

  float4 areg[8];
  float4 breg[4];

  auto load_regs = [&](int k0) {
#pragma unroll
    for (int i = 0; i < 8; ++i)
      areg[i] = *reinterpret_cast<const float4*>(
          &X[(size_t)(m0 + ar0 + i * 16) * E_ + k0 + kq]);
#pragma unroll
    for (int i = 0; i < 4; ++i)
      breg[i] = *reinterpret_cast<const float4*>(
          &W[(size_t)(k0 + br0 + i * 16) * Dh_ + d0]);
  };
  auto store_tiles = [&](int buf) {
#pragma unroll
    for (int i = 0; i < 8; ++i) {
      uint2 pk;
      pk.x = (unsigned)f2bf(areg[i].x) | ((unsigned)f2bf(areg[i].y) << 16);
      pk.y = (unsigned)f2bf(areg[i].z) | ((unsigned)f2bf(areg[i].w) << 16);
      *reinterpret_cast<uint2*>(&aT[buf][(ar0 + i * 16) * 64 + kq]) = pk;
    }
#pragma unroll
    for (int i = 0; i < 4; ++i) {
      int k = br0 + i * 16;
      bT[buf][(d0 + 0) * 64 + k] = f2bf(breg[i].x);
      bT[buf][(d0 + 1) * 64 + k] = f2bf(breg[i].y);
      bT[buf][(d0 + 2) * 64 + k] = f2bf(breg[i].z);
      bT[buf][(d0 + 3) * 64 + k] = f2bf(breg[i].w);
    }
  };

  v8f acc[4];
#pragma unroll
  for (int j = 0; j < 4; ++j) acc[j] = {};

  load_regs(0);
  store_tiles(0);

  constexpr int NC = E_ / 64;  // 16 chunks
  for (int c = 0; c < NC; ++c) {
    __syncthreads();  // buf[c&1] ready; prior reads of buf[(c+1)&1] done
    if (c + 1 < NC) load_regs((c + 1) * 64);
    const int buf = c & 1;
#pragma unroll
    for (int ks = 0; ks < 2; ++ks) {
      v16bf a = load_A(aT[buf] + wave * 16 * 64, lane, 64, ks * 32);
#pragma unroll
      for (int j = 0; j < 4; ++j) {
        v16bf bb = load_B(bT[buf], lane, 64, j * 16, ks * 32);
        acc[j] = WMMA_BF16(a, bb, acc[j]);
      }
    }
    if (c + 1 < NC) store_tiles((c + 1) & 1);
  }

  // epilogue: + bias, store bf16 to [B,H,S,Dh]
  const int hl = lane >> 4, nl = lane & 15;
#pragma unroll
  for (int j = 0; j < 4; ++j) {
    int d = j * 16 + nl;
    float bsv = bias[d];
#pragma unroll
    for (int r = 0; r < 8; ++r) {
      int m = m0 + wave * 16 + r + 8 * hl;
      int bb = m >> 11, s = m & (S_ - 1);
      size_t idx = (((size_t)bb * H_ + h) * S_ + s) * Dh_ + d;
      Out[idx] = f2bf(acc[j][r] + bsv);
    }
  }
}

// ---------------------------------------------------------------------------
// Kernel 2: flash attention. grid = (S/128, B*H), block = 256 (8 waves).
// ---------------------------------------------------------------------------
__global__ __launch_bounds__(256) void attn_kernel(
    const unsigned short* __restrict__ Qp, const unsigned short* __restrict__ Kp,
    const unsigned short* __restrict__ Vp, const unsigned char* __restrict__ mask,
    unsigned short* __restrict__ ctx) {
  __shared__ unsigned short qT[128 * 64];  // [q][dh]   A layout
  __shared__ unsigned short kT[64 * 64];   // [kv][dh]  == N-major B for QK^T
  __shared__ unsigned short vT[64 * 64];   // [dh][kv]  == N-major B for PV
  __shared__ unsigned short pT[128 * 64];  // [q][kv]   A layout for PV

  const int tid = threadIdx.x, lane = tid & 31, wave = tid >> 5;
  const int nl = lane & 15, hl = lane >> 4;
  const int q0 = blockIdx.x * 128;
  const int bh = blockIdx.y;
  const int b = bh / H_, h = bh % H_;

  const unsigned short* Qh = Qp + (size_t)bh * S_ * Dh_;
  const unsigned short* Kh = Kp + (size_t)bh * S_ * Dh_;
  const unsigned short* Vh = Vp + (size_t)bh * S_ * Dh_;

#pragma unroll
  for (int i = 0; i < 4; ++i) {  // Q tile, async to LDS
    int idx = tid + i * 256;
    int row = idx >> 3, koff = (idx & 7) * 8;
    async_b128(&qT[row * 64 + koff], &Qh[(size_t)(q0 + row) * 64 + koff]);
  }
  wait_async();
  __syncthreads();
  v16bf qf[2];
  qf[0] = load_A(qT + wave * 16 * 64, lane, 64, 0);
  qf[1] = load_A(qT + wave * 16 * 64, lane, 64, 32);

  v8f O[4];
#pragma unroll
  for (int j = 0; j < 4; ++j) O[j] = {};
  float mrow[8], lrow[8];
#pragma unroll
  for (int r = 0; r < 8; ++r) { mrow[r] = -__builtin_inff(); lrow[r] = 0.f; }

  const float scale = 0.125f;  // rsqrt(Dh=64)

  for (int kv0 = 0; kv0 < S_; kv0 += 64) {
    __syncthreads();  // previous PV reads of kT/vT done
#pragma unroll
    for (int i = 0; i < 2; ++i) {  // K chunk -> kT, async
      int idx = tid + i * 256;
      int row = idx >> 3, koff = (idx & 7) * 8;
      async_b128(&kT[row * 64 + koff], &Kh[(size_t)(kv0 + row) * 64 + koff]);
    }
#pragma unroll
    for (int i = 0; i < 2; ++i) {  // V chunk -> vT transposed [dh][kv]
      int idx = tid + i * 256;
      int kv = idx >> 3, dd = (idx & 7) * 8;
      uint4 val = *reinterpret_cast<const uint4*>(
          &Vh[(size_t)(kv0 + kv) * 64 + dd]);
      union { uint4 v; unsigned short s[8]; } u;
      u.v = val;
#pragma unroll
      for (int j = 0; j < 8; ++j) vT[(dd + j) * 64 + kv] = u.s[j];
    }
    wait_async();
    __syncthreads();

    // S = Q K^T
    v8f sa[4];
#pragma unroll
    for (int j = 0; j < 4; ++j) sa[j] = {};
#pragma unroll
    for (int kc = 0; kc < 2; ++kc) {
#pragma unroll
      for (int j = 0; j < 4; ++j) {
        v16bf bfr = load_B(kT, lane, 64, j * 16, kc * 32);
        sa[j] = WMMA_BF16(qf[kc], bfr, sa[j]);
      }
    }

    // scale + mask
    const unsigned char* mb = mask + (size_t)b * S_ * S_;
#pragma unroll
    for (int j = 0; j < 4; ++j) {
      int n = kv0 + j * 16 + nl;
#pragma unroll
      for (int r = 0; r < 8; ++r) {
        int qrow = q0 + wave * 16 + r + 8 * hl;
        unsigned char mk = mb[(size_t)qrow * S_ + n];
        sa[j][r] = mk ? sa[j][r] * scale : -1e9f;
      }
    }

    // online softmax
#pragma unroll
    for (int r = 0; r < 8; ++r) {
      float v = fmaxf(fmaxf(sa[0][r], sa[1][r]), fmaxf(sa[2][r], sa[3][r]));
#pragma unroll
      for (int off = 1; off < 16; off <<= 1) v = fmaxf(v, __shfl_xor(v, off, 32));
      float mnew = fmaxf(mrow[r], v);
      float corr = __expf(mrow[r] - mnew);
      float rs = 0.f;
#pragma unroll
      for (int j = 0; j < 4; ++j) {
        float p = __expf(sa[j][r] - mnew);
        sa[j][r] = p;
        rs += p;
      }
#pragma unroll
      for (int off = 1; off < 16; off <<= 1) rs += __shfl_xor(rs, off, 32);
      lrow[r] = lrow[r] * corr + rs;
      mrow[r] = mnew;
#pragma unroll
      for (int j = 0; j < 4; ++j) O[j][r] *= corr;
    }

    // P (C layout) -> LDS -> A layout for PV
#pragma unroll
    for (int j = 0; j < 4; ++j)
#pragma unroll
      for (int r = 0; r < 8; ++r)
        pT[(wave * 16 + r + 8 * hl) * 64 + j * 16 + nl] = f2bf(sa[j][r]);
    __syncthreads();

    // O += P V
#pragma unroll
    for (int kc = 0; kc < 2; ++kc) {
      v16bf pa = load_A(pT + wave * 16 * 64, lane, 64, kc * 32);
#pragma unroll
      for (int j = 0; j < 4; ++j) {
        v16bf bv = load_B(vT, lane, 64, j * 16, kc * 32);
        O[j] = WMMA_BF16(pa, bv, O[j]);
      }
    }
  }

  // normalize, store ctx as [B, S, H*Dh] (heads pre-concatenated), bf16
#pragma unroll
  for (int j = 0; j < 4; ++j) {
    int d = j * 16 + nl;
#pragma unroll
    for (int r = 0; r < 8; ++r) {
      int s = q0 + wave * 16 + r + 8 * hl;
      float val = O[j][r] / lrow[r];
      ctx[(((size_t)b * S_ + s) * H_ + h) * Dh_ + d] = f2bf(val);
    }
  }
}

// ---------------------------------------------------------------------------
// Kernel 3: out projection y = ctx @ Wo^T + bo. grid = (BS/128, E/64).
// Pipelined: A tile streamed async to LDS, B tile fp32->bf16 via registers.
// ---------------------------------------------------------------------------
__global__ __launch_bounds__(256) void out_proj_kernel(
    const unsigned short* __restrict__ ctx, const float* __restrict__ Wo,
    const float* __restrict__ bo, float* __restrict__ out) {
  __shared__ unsigned short aT[2][128 * 64];
  __shared__ unsigned short bT[2][64 * 64];  // N-major [n][k]

  const int tid = threadIdx.x, lane = tid & 31, wave = tid >> 5;
  const int nl = lane & 15, hl = lane >> 4;
  const int m0 = blockIdx.x * 128;
  const int n0 = blockIdx.y * 64;

  const int bn = tid >> 2, bk = (tid & 3) * 16;  // B-tile mapping
  float4 breg[4];

  auto a_async = [&](int k0, int buf) {
#pragma unroll
    for (int i = 0; i < 4; ++i) {
      int idx = tid + i * 256;
      int row = idx >> 3, koff = (idx & 7) * 8;
      async_b128(&aT[buf][row * 64 + koff],
                 &ctx[(size_t)(m0 + row) * E_ + k0 + koff]);
    }
  };
  auto b_load = [&](int k0) {
#pragma unroll
    for (int i = 0; i < 4; ++i)
      breg[i] = *reinterpret_cast<const float4*>(
          &Wo[(size_t)(n0 + bn) * E_ + k0 + bk + i * 4]);
  };
  auto b_store = [&](int buf) {
#pragma unroll
    for (int i = 0; i < 2; ++i) {
      uint4 pk;
      pk.x = (unsigned)f2bf(breg[2 * i].x) | ((unsigned)f2bf(breg[2 * i].y) << 16);
      pk.y = (unsigned)f2bf(breg[2 * i].z) | ((unsigned)f2bf(breg[2 * i].w) << 16);
      pk.z = (unsigned)f2bf(breg[2 * i + 1].x) | ((unsigned)f2bf(breg[2 * i + 1].y) << 16);
      pk.w = (unsigned)f2bf(breg[2 * i + 1].z) | ((unsigned)f2bf(breg[2 * i + 1].w) << 16);
      *reinterpret_cast<uint4*>(&bT[buf][bn * 64 + bk + i * 8]) = pk;
    }
  };

  v8f acc[4];
#pragma unroll
  for (int j = 0; j < 4; ++j) acc[j] = {};

  a_async(0, 0);
  b_load(0);
  b_store(0);
  wait_async();

  constexpr int NC = E_ / 64;
  for (int c = 0; c < NC; ++c) {
    __syncthreads();
    if (c + 1 < NC) {
      a_async((c + 1) * 64, (c + 1) & 1);
      b_load((c + 1) * 64);
    }
    const int buf = c & 1;
#pragma unroll
    for (int ks = 0; ks < 2; ++ks) {
      v16bf a = load_A(aT[buf] + wave * 16 * 64, lane, 64, ks * 32);
#pragma unroll
      for (int j = 0; j < 4; ++j) {
        v16bf bb = load_B(bT[buf], lane, 64, j * 16, ks * 32);
        acc[j] = WMMA_BF16(a, bb, acc[j]);
      }
    }
    if (c + 1 < NC) {
      b_store((c + 1) & 1);
      wait_async();
    }
  }

#pragma unroll
  for (int j = 0; j < 4; ++j) {
    int n = n0 + j * 16 + nl;
    float bsv = bo[n];
#pragma unroll
    for (int r = 0; r < 8; ++r) {
      int m = m0 + wave * 16 + r + 8 * hl;
      out[(size_t)m * E_ + n] = acc[j][r] + bsv;
    }
  }
}

// ---------------------------------------------------------------------------
extern "C" void kernel_launch(void* const* d_in, const int* in_sizes, int n_in,
                              void* d_out, int out_size, void* d_ws,
                              size_t ws_size, hipStream_t stream) {
  const float* qx = (const float*)d_in[0];
  const float* kx = (const float*)d_in[1];
  const float* vx = (const float*)d_in[2];
  const unsigned char* mask = (const unsigned char*)d_in[3];
  const float* Wq = (const float*)d_in[4];
  const float* bq = (const float*)d_in[5];
  const float* Wk = (const float*)d_in[6];
  const float* bk = (const float*)d_in[7];
  const float* Wv = (const float*)d_in[8];
  const float* bv = (const float*)d_in[9];
  const float* Wo = (const float*)d_in[10];
  const float* bo = (const float*)d_in[11];

  unsigned short* ws = (unsigned short*)d_ws;
  const size_t PROJ = (size_t)B_ * H_ * S_ * Dh_;  // 8,388,608 elems (16 MB bf16)
  unsigned short* Qp = ws;
  unsigned short* Kp = ws + PROJ;
  unsigned short* Vp = ws + 2 * PROJ;
  unsigned short* ctx = ws + 3 * PROJ;  // total ws use = 64 MB

  qkv_proj_kernel<<<dim3(BS_ / 128, 3 * H_), 256, 0, stream>>>(
      qx, kx, vx, Wq, Wk, Wv, bq, bk, bv, Qp, Kp, Vp);
  attn_kernel<<<dim3(S_ / 128, B_ * H_), 256, 0, stream>>>(Qp, Kp, Vp, mask, ctx);
  out_proj_kernel<<<dim3(BS_ / 128, E_ / 64), 256, 0, stream>>>(
      ctx, Wo, bo, (float*)d_out);
}
